// DeformConv_5858335392303
// MI455X (gfx1250) — matrix operand order repaired and checked by reading, not compiled, not run on hardware
//
#include <hip/hip_runtime.h>
#include <hip/hip_bf16.h>

typedef unsigned short u16;
typedef unsigned int   u32;
typedef __attribute__((ext_vector_type(16))) __bf16 v16bf;
typedef __attribute__((ext_vector_type(8)))  float  v8f;

// ---------------- problem constants ----------------
static constexpr int IMGS = 8;          // B*F
static constexpr int CH   = 64;         // NF
static constexpr int HH   = 96, WW = 96;
static constexpr int HW   = HH * WW;    // 9216
static constexpr int HP   = 98, WP = 98;// padded spatial
static constexpr int C1   = 128;        // concat channels
static constexpr int DG   = 8, KKT = 9; // deform groups, taps
static constexpr int XP2  = WW / 32;    // 3 pixel-tile-pairs per row
static constexpr int PT2  = HH * XP2;   // 288 blocks per image (2 N-tiles each)

__device__ __forceinline__ u16 to_bf16(float f) {
  u32 u = __builtin_bit_cast(u32, f);
  u += 0x7FFFu + ((u >> 16) & 1u);      // round-to-nearest-even
  return (u16)(u >> 16);
}

__device__ __forceinline__ v8f zero8() {
  v8f z = {0.f,0.f,0.f,0.f,0.f,0.f,0.f,0.f};
  return z;
}

union FragBF { uint4 q[2]; v16bf v; };

__device__ __forceinline__ void load_frag(FragBF& f, const u16* p) {
  f.q[0] = ((const uint4*)p)[0];
  f.q[1] = ((const uint4*)p)[1];
}

__device__ __forceinline__ v8f wmma_bf16(const FragBF& a, const FragBF& b, v8f c) {
  return __builtin_amdgcn_wmma_f32_16x16x32_bf16(false, a.v, false, b.v,
                                                 (short)0, c, false, false);
}

// ---------------- weight packing into A-fragment order ----------------
// K-order: k = tap*Cin + ci ; packed[(((mt*nKt+kt)*32+lane)*16)+j]
// A 16x32 bf16 packing used here (contiguous k, B loads match): lane<16 -> rows 0..15, K {0..15}; lane>=16 -> K {16..31}
__global__ void pack_w(const float* __restrict__ w, int Cout, int Cin,
                       int nMt, int nKt, u16* __restrict__ dst) {
  int tid = blockIdx.x * blockDim.x + threadIdx.x;
  int total = nMt * nKt * 512;
  if (tid >= total) return;
  int j     = tid & 15;
  int lane  = (tid >> 4) & 31;
  int mk    = tid >> 9;
  int kt    = mk % nKt;
  int mt    = mk / nKt;
  int khalf = lane >> 4;
  int klocal = khalf * 16 + j;
  int k  = kt * 32 + klocal;
  int co = mt * 16 + (lane & 15);
  int tap = k / Cin, ci = k % Cin;
  int ky = tap / 3, kx = tap % 3;
  float v = 0.f;
  if (co < Cout) v = w[((co * Cin + ci) * 3 + ky) * 3 + kx];
  dst[tid] = to_bf16(v);
}

// ---------------- stage concat(x,r) -> padded channel-last bf16 ----------------
__global__ void stage_xr(const float* __restrict__ x, const float* __restrict__ r,
                         u16* __restrict__ dst) {
  int tid = blockIdx.x * blockDim.x + threadIdx.x;
  if (tid >= HP * WP * C1) return;
  int ci = tid % C1;
  int pp = tid / C1;
  int xp = pp % WP, yp = pp / WP;
  float v = 0.f;
  if (yp >= 1 && yp <= HH && xp >= 1 && xp <= WW) {
    const float* s = (ci < CH) ? x : r;
    v = s[(ci & 63) * HW + (yp - 1) * WW + (xp - 1)];
  }
  dst[tid] = to_bf16(v);
}

__global__ void fill0(u32* p, int n) {
  int i = blockIdx.x * blockDim.x + threadIdx.x;
  if (i < n) p[i] = 0u;
}

// ---------------- implicit-GEMM 3x3 conv via bf16 WMMA ----------------
// One wave: 32 pixels (two 16-pixel N-tiles) x 4x16 out-channel tiles.
// A fragments reused across both N-tiles; all WMMAs unconditional.
template <int CIN, bool LRELU, bool OUT_BF16>
__global__ __launch_bounds__(32) void conv_gemm(
    const u16* __restrict__ inPad,
    const u16* __restrict__ wpack,
    const float* __restrict__ bias, int CoutReal,
    u16* __restrict__ outPad,
    float* __restrict__ outF32) {
  constexpr int NKT = CIN * 9 / 32;   // K tiles
  constexpr int TPT = CIN / 32;       // K tiles per tap
  int lane = threadIdx.x;
  int t = blockIdx.x;
  int y = t / XP2;
  int xb = (t % XP2) * 32;
  int mtBase = blockIdx.y * 4;
  int p = lane & 15;
  int khalf = lane >> 4;
  int px0 = xb + p;          // N-tile 0 pixel
  // N-tile 1 pixel = px0 + 16

  v8f a0n0 = zero8(), a1n0 = zero8(), a2n0 = zero8(), a3n0 = zero8();
  v8f a0n1 = zero8(), a1n1 = zero8(), a2n1 = zero8(), a3n1 = zero8();

  // per-wave A base: lane-fixed offset, mt stride = NKT*512, kt stride = 512
  const u16* aBase = wpack + (size_t)mtBase * NKT * 512 + lane * 16;

#pragma unroll
  for (int tap = 0; tap < 9; ++tap) {
    const int ki = tap / 3, kj = tap % 3;
    const u16* brow0 = inPad + (((y + ki) * WP + (px0 + kj)) * CIN + khalf * 16);
    const u16* brow1 = brow0 + 16 * CIN;
    __builtin_prefetch(brow0 + CIN, 0, 1);   // global_prefetch_b8
#pragma unroll
    for (int rem = 0; rem < TPT; ++rem) {
      const int kt = tap * TPT + rem;
      FragBF B0, B1;
      load_frag(B0, brow0 + rem * 32);
      load_frag(B1, brow1 + rem * 32);
      const u16* ap = aBase + (size_t)kt * 512;
      FragBF A0, A1, A2, A3;
      load_frag(A0, ap);
      load_frag(A1, ap + (size_t)NKT * 512);
      load_frag(A2, ap + (size_t)2 * NKT * 512);
      load_frag(A3, ap + (size_t)3 * NKT * 512);
      a0n0 = wmma_bf16(A0, B0, a0n0);
      a0n1 = wmma_bf16(A0, B1, a0n1);
      a1n0 = wmma_bf16(A1, B0, a1n0);
      a1n1 = wmma_bf16(A1, B1, a1n1);
      a2n0 = wmma_bf16(A2, B0, a2n0);
      a2n1 = wmma_bf16(A2, B1, a2n1);
      a3n0 = wmma_bf16(A3, B0, a3n0);
      a3n1 = wmma_bf16(A3, B1, a3n1);
    }
  }

  // epilogue (both N-tiles)
#pragma unroll
  for (int nt = 0; nt < 2; ++nt) {
    const int px = px0 + nt * 16;
    const int outPix = (y + 1) * WP + (px + 1);
    const int planar = y * WW + px;
#pragma unroll
    for (int mt = 0; mt < 4; ++mt) {
      v8f a;
      if (nt == 0) a = (mt == 0) ? a0n0 : (mt == 1) ? a1n0 : (mt == 2) ? a2n0 : a3n0;
      else         a = (mt == 0) ? a0n1 : (mt == 1) ? a1n1 : (mt == 2) ? a2n1 : a3n1;
#pragma unroll
      for (int i = 0; i < 8; ++i) {
        int co = (mtBase + mt) * 16 + khalf * 8 + i;
        if (co < CoutReal) {
          float f = a[i] + bias[co];
          if (LRELU) f = (f >= 0.f) ? f : 0.1f * f;
          if (OUT_BF16) outPad[outPix * CH + co] = to_bf16(f);
          else          outF32[(size_t)co * HW + planar] = f;
        }
      }
    }
  }
}

// ---------------- DCNv2 gather: V[p][tap*64+ci] = mask * bilinear(x) ----------------
__global__ void dcn_gather(const float* __restrict__ xsrc, const float* __restrict__ om,
                           u16* __restrict__ V) {
  int tid = blockIdx.x * blockDim.x + threadIdx.x;
  if (tid >= HW * DG * KKT) return;
  int pg = tid % HW;
  int kg = tid / HW;          // 0..71
  int k  = kg / DG;           // tap 0..8
  int g  = kg % DG;           // group 0..7
  int y = pg / WW, x = pg % WW;
  int ki = k / 3, kj = k % 3;
  int chy = g * KKT + k;
  float offy = om[(size_t)chy * HW + pg];
  float offx = om[(size_t)(72 + chy) * HW + pg];
  float mraw = om[(size_t)(144 + chy) * HW + pg];
  float m = 1.f / (1.f + __expf(-mraw));

  float py = (float)y + (float)(ki - 1) + offy;
  float px = (float)x + (float)(kj - 1) + offx;
  float y0f = floorf(py), x0f = floorf(px);
  float wy1 = py - y0f, wx1 = px - x0f;
  float wy0 = 1.f - wy1, wx0 = 1.f - wx1;
  bool vy0 = (y0f >= 0.f) && (y0f <= (float)(HH - 1));
  bool vy1 = (y0f + 1.f >= 0.f) && (y0f + 1.f <= (float)(HH - 1));
  bool vx0 = (x0f >= 0.f) && (x0f <= (float)(WW - 1));
  bool vx1 = (x0f + 1.f >= 0.f) && (x0f + 1.f <= (float)(WW - 1));
  int y0 = min(max((int)y0f, 0), HH - 1);
  int y1 = min(max((int)y0f + 1, 0), HH - 1);
  int x0 = min(max((int)x0f, 0), WW - 1);
  int x1 = min(max((int)x0f + 1, 0), WW - 1);
  float w00 = wy0 * wx0 * ((vy0 && vx0) ? 1.f : 0.f) * m;
  float w01 = wy0 * wx1 * ((vy0 && vx1) ? 1.f : 0.f) * m;
  float w10 = wy1 * wx0 * ((vy1 && vx0) ? 1.f : 0.f) * m;
  float w11 = wy1 * wx1 * ((vy1 && vx1) ? 1.f : 0.f) * m;
  int i00 = y0 * WW + x0, i01 = y0 * WW + x1;
  int i10 = y1 * WW + x0, i11 = y1 * WW + x1;
#pragma unroll
  for (int c = 0; c < 8; ++c) {
    int ci = g * 8 + c;
    const float* s = xsrc + (size_t)ci * HW;
    float v = w00 * s[i00] + w01 * s[i01] + w10 * s[i10] + w11 * s[i11];
    V[(size_t)pg * 576 + k * 64 + ci] = to_bf16(v);
  }
}

// ---------------- DCN einsum: out[co,p] = W_dcn x V + b ----------------
// One wave: 32 pixels x 64 out channels, A reused across both N-tiles.
__global__ __launch_bounds__(32) void dcn_gemm(
    const u16* __restrict__ V, const u16* __restrict__ wpack,
    const float* __restrict__ bias, float* __restrict__ out) {
  constexpr int NKT = 18;
  int lane = threadIdx.x;
  int t = blockIdx.x;
  int y = t / XP2;
  int xb = (t % XP2) * 32;
  int p = lane & 15;
  int khalf = lane >> 4;
  int pix0 = y * WW + xb + p;

  v8f a0n0 = zero8(), a1n0 = zero8(), a2n0 = zero8(), a3n0 = zero8();
  v8f a0n1 = zero8(), a1n1 = zero8(), a2n1 = zero8(), a3n1 = zero8();

  const u16* brow0 = V + (size_t)pix0 * 576 + khalf * 16;
  const u16* brow1 = brow0 + (size_t)16 * 576;
  const u16* aBase = wpack + lane * 16;

#pragma unroll
  for (int kt = 0; kt < NKT; ++kt) {
    FragBF B0, B1;
    load_frag(B0, brow0 + kt * 32);
    load_frag(B1, brow1 + kt * 32);
    const u16* ap = aBase + (size_t)kt * 512;
    FragBF A0, A1, A2, A3;
    load_frag(A0, ap);
    load_frag(A1, ap + (size_t)NKT * 512);
    load_frag(A2, ap + (size_t)2 * NKT * 512);
    load_frag(A3, ap + (size_t)3 * NKT * 512);
    a0n0 = wmma_bf16(A0, B0, a0n0);
    a0n1 = wmma_bf16(A0, B1, a0n1);
    a1n0 = wmma_bf16(A1, B0, a1n0);
    a1n1 = wmma_bf16(A1, B1, a1n1);
    a2n0 = wmma_bf16(A2, B0, a2n0);
    a2n1 = wmma_bf16(A2, B1, a2n1);
    a3n0 = wmma_bf16(A3, B0, a3n0);
    a3n1 = wmma_bf16(A3, B1, a3n1);
  }

#pragma unroll
  for (int nt = 0; nt < 2; ++nt) {
    int pix = pix0 + nt * 16;
#pragma unroll
    for (int mt = 0; mt < 4; ++mt) {
      v8f a;
      if (nt == 0) a = (mt == 0) ? a0n0 : (mt == 1) ? a1n0 : (mt == 2) ? a2n0 : a3n0;
      else         a = (mt == 0) ? a0n1 : (mt == 1) ? a1n1 : (mt == 2) ? a2n1 : a3n1;
#pragma unroll
      for (int i = 0; i < 8; ++i) {
        int co = mt * 16 + khalf * 8 + i;
        out[(size_t)co * HW + pix] = a[i] + bias[co];
      }
    }
  }
}

// ---------------- host launch ----------------
extern "C" void kernel_launch(void* const* d_in, const int* in_sizes, int n_in,
                              void* d_out, int out_size, void* d_ws, size_t ws_size,
                              hipStream_t stream) {
  const float* nbr   = (const float*)d_in[0];
  const float* ref   = (const float*)d_in[1];
  const float* w1    = (const float*)d_in[2];
  const float* b1    = (const float*)d_in[3];
  const float* w2    = (const float*)d_in[4];
  const float* b2    = (const float*)d_in[5];
  const float* w_off = (const float*)d_in[6];
  const float* b_off = (const float*)d_in[7];
  const float* w_dcn = (const float*)d_in[8];
  const float* b_dcn = (const float*)d_in[9];
  float* out = (float*)d_out;

  // workspace carve (256B aligned)
  char* ws = (char*)d_ws;
  auto carve = [&](size_t bytes) -> char* {
    char* r = ws;
    ws += (bytes + 255) & ~(size_t)255;
    return r;
  };
  u16* w1p = (u16*)carve((size_t)4  * 36 * 512 * 2);
  u16* w2p = (u16*)carve((size_t)4  * 18 * 512 * 2);
  u16* wop = (u16*)carve((size_t)16 * 18 * 512 * 2);  // padded to 16 M-tiles (zeros)
  u16* wdp = (u16*)carve((size_t)4  * 18 * 512 * 2);
  u16* xrp = (u16*)carve((size_t)HP * WP * C1 * 2);
  u16* o1p = (u16*)carve((size_t)HP * WP * CH * 2);
  u16* o2p = (u16*)carve((size_t)HP * WP * CH * 2);
  float* om = (float*)carve((size_t)216 * HW * 4);
  u16* V   = (u16*)carve((size_t)HW * 576 * 2);

  // pack weights (bf16 A-fragment order)
  pack_w<<<(4 * 36 * 512 + 255) / 256, 256, 0, stream>>>(w1, 64, 128, 4, 36, w1p);
  pack_w<<<(4 * 18 * 512 + 255) / 256, 256, 0, stream>>>(w2, 64, 64, 4, 18, w2p);
  pack_w<<<(16 * 18 * 512 + 255) / 256, 256, 0, stream>>>(w_off, 216, 64, 16, 18, wop);
  pack_w<<<(4 * 18 * 512 + 255) / 256, 256, 0, stream>>>(w_dcn, 64, 64, 4, 18, wdp);

  // zero padded activation buffers (borders must be 0; interiors rewritten per image)
  int padWords = (HP * WP * CH * 2) / 4;
  fill0<<<(padWords + 255) / 256, 256, 0, stream>>>((u32*)o1p, padWords);
  fill0<<<(padWords + 255) / 256, 256, 0, stream>>>((u32*)o2p, padWords);

  for (int n = 0; n < IMGS; ++n) {
    const float* xn = nbr + (size_t)n * CH * HW;
    const float* rn = ref + (size_t)n * CH * HW;
    float* outn = out + (size_t)n * CH * HW;

    stage_xr<<<(HP * WP * C1 + 255) / 256, 256, 0, stream>>>(xn, rn, xrp);

    // conv1: 128->64, lrelu, bf16 padded out
    conv_gemm<128, true, true><<<dim3(PT2, 1), 32, 0, stream>>>(
        xrp, w1p, b1, 64, o1p, nullptr);
    // conv2: 64->64, lrelu
    conv_gemm<64, true, true><<<dim3(PT2, 1), 32, 0, stream>>>(
        o1p, w2p, b2, 64, o2p, nullptr);
    // conv_off: 64->216 (weights padded to 256 outs; 4 chunks x 4 M-tiles), f32 planar out
    conv_gemm<64, false, false><<<dim3(PT2, 4), 32, 0, stream>>>(
        o2p, wop, b_off, 216, nullptr, om);

    // modulated deformable gather -> V (bf16)
    dcn_gather<<<(HW * DG * KKT + 255) / 256, 256, 0, stream>>>(xn, om, V);
    // DCN einsum via WMMA
    dcn_gemm<<<PT2, 32, 0, stream>>>(V, wdp, b_dcn, outn);
  }
}
